// Attention_66374424592893
// MI455X (gfx1250) — matrix operand order repaired and checked by reading, not compiled, not run on hardware
//
#include <hip/hip_runtime.h>
#include <hip/hip_bf16.h>

// ---------------------------------------------------------------------------
// Fused attention pipeline for MI455X (gfx1250, wave32).
// B=4, L=1024, D=768, H=12, DH=64.
//
// All five einsums run as NT-GEMM  C[M,N] = A[M,K] * Bt[N,K]^T  on
// v_wmma_f32_16x16x32_bf16, with tiles staged into double-buffered LDS by the
// CDNA5 async copy engine (global_load_async_to_lds_b128 / ASYNCcnt) so the
// copy of K-step i+1 overlaps the WMMAs of K-step i.
// ---------------------------------------------------------------------------

typedef __attribute__((ext_vector_type(16))) __bf16 v16bf;
typedef __attribute__((ext_vector_type(8)))  float  v8f;

#define B_  4
#define L_  1024
#define D_  768
#define H_  12
#define DH_ 64

// ------------------------- CDNA5 async-copy helpers ------------------------
__device__ __forceinline__ void async_b128_to_lds(unsigned lds_off, const void* gaddr) {
    // GLOBAL_LOAD_ASYNC_TO_LDS_B128: VDST = per-lane LDS byte address,
    // VADDR = per-lane 64-bit global address (GV mode). Tracked by ASYNCcnt.
    asm volatile("global_load_async_to_lds_b128 %0, %1, off"
                 :: "v"(lds_off), "v"(gaddr) : "memory");
}

__device__ __forceinline__ void wait_async_zero() {
#if __has_builtin(__builtin_amdgcn_s_wait_asynccnt)
    __builtin_amdgcn_s_wait_asynccnt(0);
#else
    asm volatile("s_wait_asynccnt 0x0" ::: "memory");
#endif
}

// LDS flat addresses are {SHARED_BASE, offset[31:0]} -> low 32 bits = LDS byte offset.
__device__ __forceinline__ unsigned lds_off(const void* p) {
    return (unsigned)(unsigned long long)p;
}

// ------------------------------ f32 -> bf16 --------------------------------
__global__ __launch_bounds__(256)
void cvt_f32_to_bf16(const float* __restrict__ in, __bf16* __restrict__ out, long n) {
    long i = (long)blockIdx.x * 256 + threadIdx.x;
    long stride = (long)gridDim.x * 256;
    for (; i < n; i += stride) out[i] = (__bf16)in[i];
}

// ------------------------------ NT GEMM ------------------------------------
// Block tile 128(M) x 64(N), K-step 32. 256 threads = 8 waves in a 4x2 grid,
// each wave owns a 32x32 tile = 2x2 v_wmma_f32_16x16x32_bf16 accumulators.
constexpr int BM = 128;
constexpr int BN = 64;
constexpr int BK = 32;
constexpr int LDP = BK + 8;   // 80B LDS rows: 16B-aligned chunks, no bank conflicts

template<bool OUT_BF16>
__global__ __launch_bounds__(256)
void gemm_nt_bf16(const __bf16* __restrict__ A,  // [batch] M x K row-major
                  const __bf16* __restrict__ Bt, // [batch] N x K row-major (col-major B)
                  void* __restrict__ C,
                  int M, int N, int K,
                  long sAb, long sAh, long sBb, long sBh, long sCb, long sCh,
                  int ldc, int nH) {
    __shared__ __bf16 Als[2][BM][LDP];
    __shared__ __bf16 Bls[2][BN][LDP];

    const int bz = blockIdx.z;
    const int bb = bz / nH, hh = bz % nH;
    A  += (long)bb * sAb + (long)hh * sAh;
    Bt += (long)bb * sBb + (long)hh * sBh;
    const size_t coff = (size_t)bb * sCb + (size_t)hh * sCh;

    const int tid  = threadIdx.x;
    const int wave = tid >> 5;
    const int lane = tid & 31;
    const int wm   = wave >> 1;      // 0..3  (M direction)
    const int wn   = wave & 1;       // 0..1  (N direction)
    const int lrow = lane & 15;
    const int g    = lane >> 4;      // lane group per ISA fragment layout

    const int m_blk = blockIdx.y * BM;
    const int n_blk = blockIdx.x * BN;

    // global->LDS staging assignment (A: 2x16B/thread, B: 1x16B/thread)
    const int ar = tid >> 1;            // 0..127
    const int ah = (tid & 1) * 16;      // col 0 or 16
    const int br = tid >> 2;            // 0..63
    const int bc = (tid & 3) * 8;       // col 0/8/16/24

    // No zero-fill for M-overhang rows: garbage A rows only feed C rows >= M,
    // which the store predicate drops; reads stay inside the workspace.
    const __bf16* aptr = A  + (size_t)(m_blk + ar) * K + ah;
    const __bf16* bptr = Bt + (size_t)(n_blk + br) * K + bc;

    const unsigned la0 = lds_off(&Als[0][ar][ah]);
    const unsigned la1 = lds_off(&Als[0][ar][ah + 8]);
    const unsigned lb0 = lds_off(&Bls[0][br][bc]);
    const unsigned dA  = (unsigned)(sizeof(__bf16) * BM * LDP);  // Als buffer stride
    const unsigned dB  = (unsigned)(sizeof(__bf16) * BN * LDP);  // Bls buffer stride

    auto stage = [&](int buf, int k0) {
        async_b128_to_lds(la0 + buf * dA, aptr + k0);
        async_b128_to_lds(la1 + buf * dA, aptr + k0 + 8);
        async_b128_to_lds(lb0 + buf * dB, bptr + k0);
    };

    v8f acc[2][2] = {};

    stage(0, 0);
    int buf = 0;
    for (int k0 = 0; k0 < K; k0 += BK, buf ^= 1) {
        wait_async_zero();      // stage(k0) landed in LDS
        __syncthreads();        // publish it; fence last iter's reads of buf^1

        if (k0 + BK < K) stage(buf ^ 1, k0 + BK);        // overlaps the WMMAs below
        if (k0 + 2 * BK < K) {                           // warm L2 two tiles ahead
            __builtin_prefetch(aptr + k0 + 2 * BK, 0, 1);
            __builtin_prefetch(bptr + k0 + 2 * BK, 0, 1);
        }

        // ---- fragments per documented CDNA5 WMMA VGPR layouts ----
        // A 16x32: lane (g,lrow) holds row lrow, K = {g*8+0..7, 16+g*8+0..7}
        v16bf afrag[2];
#pragma unroll
        for (int mi = 0; mi < 2; ++mi) {
            const int r = wm * 32 + mi * 16 + lrow;
            uint4* ap = reinterpret_cast<uint4*>(&afrag[mi]);
            ap[0] = *reinterpret_cast<const uint4*>(&Als[buf][r][g * 8]);
            ap[1] = *reinterpret_cast<const uint4*>(&Als[buf][r][16 + g * 8]);
        }
        // B 32x16: lane (g,lrow) holds column lrow, K = {g*8+0..7, 16+g*8+0..7}
        v16bf bfrag[2];
#pragma unroll
        for (int ni = 0; ni < 2; ++ni) {
            const int n = wn * 32 + ni * 16 + lrow;
            uint4* bp = reinterpret_cast<uint4*>(&bfrag[ni]);
            bp[0] = *reinterpret_cast<const uint4*>(&Bls[buf][n][g * 8]);
            bp[1] = *reinterpret_cast<const uint4*>(&Bls[buf][n][16 + g * 8]);
        }

#pragma unroll
        for (int mi = 0; mi < 2; ++mi)
#pragma unroll
            for (int ni = 0; ni < 2; ++ni)
                acc[mi][ni] = __builtin_amdgcn_wmma_f32_16x16x32_bf16(
                    false, afrag[mi], false, bfrag[ni],
                    (short)0, acc[mi][ni], false, false);
    }

    // ---- store C: VGPR r holds (M = r + 8*g, N = lrow) of each 16x16 tile ----
#pragma unroll
    for (int mi = 0; mi < 2; ++mi) {
#pragma unroll
        for (int ni = 0; ni < 2; ++ni) {
            const int row0 = m_blk + wm * 32 + mi * 16 + g * 8;
            const int col  = n_blk + wn * 32 + ni * 16 + lrow;
#pragma unroll
            for (int r = 0; r < 8; ++r) {
                const int row = row0 + r;
                if (row < M) {
                    const float val = acc[mi][ni][r];
                    if constexpr (OUT_BF16)
                        reinterpret_cast<__bf16*>(C)[coff + (size_t)row * ldc + col] = (__bf16)val;
                    else
                        reinterpret_cast<float*>(C)[coff + (size_t)row * ldc + col] = val;
                }
            }
        }
    }
}

// --------------------------- row softmax (in place) ------------------------
// One 256-thread block per row of 1024 bf16 logits; f32 math internally.
__global__ __launch_bounds__(256)
void softmax_row_bf16(__bf16* __restrict__ S) {
    __shared__ float red[8];
    __bf16* p = S + (size_t)blockIdx.x * L_;
    const int tid  = threadIdx.x;
    const int wave = tid >> 5;
    const int lane = tid & 31;

    float v[4];
#pragma unroll
    for (int j = 0; j < 4; ++j) v[j] = (float)p[tid + j * 256];

    float m = fmaxf(fmaxf(v[0], v[1]), fmaxf(v[2], v[3]));
#pragma unroll
    for (int off = 16; off > 0; off >>= 1) m = fmaxf(m, __shfl_xor(m, off));
    if (lane == 0) red[wave] = m;
    __syncthreads();
    if (tid == 0) {
        float mm = red[0];
        for (int i = 1; i < 8; ++i) mm = fmaxf(mm, red[i]);
        red[0] = mm;
    }
    __syncthreads();
    m = red[0];
    __syncthreads();

    float s = 0.f;
#pragma unroll
    for (int j = 0; j < 4; ++j) { v[j] = __expf(v[j] - m); s += v[j]; }
#pragma unroll
    for (int off = 16; off > 0; off >>= 1) s += __shfl_xor(s, off);
    if (lane == 0) red[wave] = s;
    __syncthreads();
    if (tid == 0) {
        float ss = 0.f;
        for (int i = 0; i < 8; ++i) ss += red[i];
        red[0] = ss;
    }
    __syncthreads();
    const float inv = 1.0f / red[0];
#pragma unroll
    for (int j = 0; j < 4; ++j) p[tid + j * 256] = (__bf16)(v[j] * inv);
}

// ---------------------------------------------------------------------------
extern "C" void kernel_launch(void* const* d_in, const int* in_sizes, int n_in,
                              void* d_out, int out_size, void* d_ws, size_t ws_size,
                              hipStream_t stream) {
    const float* x = (const float*)d_in[0];   // [B,L,D]
    const float* k = (const float*)d_in[1];   // [H,D,D]
    const float* q = (const float*)d_in[2];   // [H,D,D]
    const float* v = (const float*)d_in[3];   // [H,DH,D]
    float* out = (float*)d_out;               // [B,L,D] f32

    // workspace carve-out (~294 MB total)
    char* ws = (char*)d_ws;
    auto carve = [&](size_t bytes) -> char* {
        char* p = ws;
        ws += (bytes + 255) & ~(size_t)255;
        return p;
    };
    __bf16* xb = (__bf16*)carve((size_t)B_ * L_ * D_ * 2);
    __bf16* kb = (__bf16*)carve((size_t)H_ * D_ * D_ * 2);
    __bf16* qb = (__bf16*)carve((size_t)H_ * D_ * D_ * 2);
    __bf16* vb = (__bf16*)carve((size_t)H_ * DH_ * D_ * 2);
    __bf16* Qt = (__bf16*)carve((size_t)B_ * H_ * L_ * D_ * 2);   // [b,h,l,c]
    __bf16* Kt = (__bf16*)carve((size_t)B_ * H_ * L_ * D_ * 2);   // [b,h,l,c]
    __bf16* Vp = (__bf16*)carve((size_t)B_ * H_ * DH_ * L_ * 2);  // [b,h,c,l]
    __bf16* S  = (__bf16*)carve((size_t)B_ * H_ * L_ * L_ * 2);   // [b,h,l,m]

    // 1) convert inputs to bf16
    cvt_f32_to_bf16<<<2048, 256, 0, stream>>>(x, xb, (long)B_ * L_ * D_);
    cvt_f32_to_bf16<<<2048, 256, 0, stream>>>(k, kb, (long)H_ * D_ * D_);
    cvt_f32_to_bf16<<<2048, 256, 0, stream>>>(q, qb, (long)H_ * D_ * D_);
    cvt_f32_to_bf16<<<1024, 256, 0, stream>>>(v, vb, (long)H_ * DH_ * D_);

    const dim3 blk(256);
    const int BH = B_ * H_;

    // 2) Qt[b,h,l,c] = sum_d x[b,l,d] * q[h,c,d]   (M=L, N=D, K=D)
    gemm_nt_bf16<true><<<dim3(D_ / BN, L_ / BM, BH), blk, 0, stream>>>(
        xb, qb, (void*)Qt, L_, D_, D_,
        (long)L_ * D_, 0, 0, (long)D_ * D_,
        (long)H_ * L_ * D_, (long)L_ * D_, D_, H_);
    // 3) Kt[b,h,l,c]
    gemm_nt_bf16<true><<<dim3(D_ / BN, L_ / BM, BH), blk, 0, stream>>>(
        xb, kb, (void*)Kt, L_, D_, D_,
        (long)L_ * D_, 0, 0, (long)D_ * D_,
        (long)H_ * L_ * D_, (long)L_ * D_, D_, H_);
    // 4) Vp[b,h,c,l] = sum_d v[h,c,d] * x[b,l,d]   (M=DH, N=L, K=D)
    gemm_nt_bf16<true><<<dim3(L_ / BN, (DH_ + BM - 1) / BM, BH), blk, 0, stream>>>(
        vb, xb, (void*)Vp, DH_, L_, D_,
        0, (long)DH_ * D_, (long)L_ * D_, 0,
        (long)H_ * DH_ * L_, (long)DH_ * L_, L_, H_);
    // 5) S[l,m] = sum_c Qt[l,c] * Kt[m,c]          (M=L, N=L, K=D)
    gemm_nt_bf16<true><<<dim3(L_ / BN, L_ / BM, BH), blk, 0, stream>>>(
        Qt, Kt, (void*)S, L_, L_, D_,
        (long)H_ * L_ * D_, (long)L_ * D_, (long)H_ * L_ * D_, (long)L_ * D_,
        (long)H_ * L_ * L_, (long)L_ * L_, L_, H_);
    // 6) softmax over m, in place (one block per row)
    softmax_row_bf16<<<BH * L_, 256, 0, stream>>>(S);
    // 7) out[b,l,h*DH+c] = sum_m S[l,m] * Vp[c,m]  (M=L, N=DH, K=L), f32 out
    gemm_nt_bf16<false><<<dim3(DH_ / BN, L_ / BM, BH), blk, 0, stream>>>(
        S, Vp, (void*)out, L_, DH_, L_,
        (long)H_ * L_ * L_, (long)L_ * L_, (long)H_ * DH_ * L_, (long)DH_ * L_,
        (long)L_ * D_, (long)DH_, D_, H_);
}
